// LSTMModel_82978768159511
// MI455X (gfx1250) — compile-verified, matrix operand order checked by reading, not run on hardware
//
#include <hip/hip_runtime.h>
#include <math.h>

typedef __attribute__((ext_vector_type(16))) _Float16 v16h;
typedef __attribute__((ext_vector_type(8)))  _Float16 v8h;
typedef __attribute__((ext_vector_type(8)))  float    v8f;

// ---------------------------------------------------------------------------
// Generic fused Conv1d (+BN/ReLU variants)
// mode 0: ReLU(BN(conv)) ; mode 1: BN(ReLU(conv)) ; mode 2: ReLU(conv)
// x: [T, Cin, Lin]  w: [Cout, Cin, K]  y: [T, Cout, Lout]
// ---------------------------------------------------------------------------
__global__ __launch_bounds__(256) void conv1d_fused(
    const float* __restrict__ x, const float* __restrict__ w,
    const float* __restrict__ bias,
    const float* __restrict__ bng, const float* __restrict__ bnb,
    const float* __restrict__ bnm, const float* __restrict__ bnv,
    float* __restrict__ y,
    int T, int Cin, int Lin, int Cout, int Kc, int stride, int Lout, int mode)
{
    int idx = blockIdx.x * blockDim.x + threadIdx.x;
    int total = T * Cout * Lout;
    if (idx >= total) return;
    int l  = idx % Lout;
    int tc = idx / Lout;
    int co = tc % Cout;
    int t  = tc / Cout;

    float acc = bias[co];
    const float* xp = x + (size_t)t * Cin * Lin + l * stride;
    const float* wp = w + (size_t)co * Cin * Kc;
    for (int ci = 0; ci < Cin; ci++) {
        const float* xr = xp + (size_t)ci * Lin;
        const float* wr = wp + ci * Kc;
        #pragma unroll 4
        for (int k = 0; k < Kc; k++) acc += xr[k] * wr[k];
    }
    if (mode == 0) {
        float s = bng[co] * rsqrtf(bnv[co] + 1e-5f);
        acc = (acc - bnm[co]) * s + bnb[co];
        acc = fmaxf(acc, 0.f);
    } else if (mode == 1) {
        acc = fmaxf(acc, 0.f);
        float s = bng[co] * rsqrtf(bnv[co] + 1e-5f);
        acc = (acc - bnm[co]) * s + bnb[co];
    } else {
        acc = fmaxf(acc, 0.f);
    }
    y[idx] = acc;
}

// AvgPool1d(k=4, s=2) + channel-major flatten; emits f16 (GEMM operand)
__global__ __launch_bounds__(256) void avgpool_flatten_f16(
    const float* __restrict__ x, _Float16* __restrict__ y,
    int T, int C, int L, int Lp)
{
    int idx = blockIdx.x * blockDim.x + threadIdx.x;
    int total = T * C * Lp;
    if (idx >= total) return;
    int lp = idx % Lp;
    int tc = idx / Lp;
    int c  = tc % C;
    int t  = tc / C;
    const float* xp = x + ((size_t)t * C + c) * L + 2 * lp;
    float s = 0.25f * (xp[0] + xp[1] + xp[2] + xp[3]);
    y[(size_t)t * (C * Lp) + c * Lp + lp] = (_Float16)s;
}

// Forces [3, T, 125] -> [T, 3, 125]
__global__ __launch_bounds__(256) void transpose_forces(
    const float* __restrict__ in, float* __restrict__ out, int T, int L)
{
    int idx = blockIdx.x * blockDim.x + threadIdx.x;
    int total = 3 * T * L;
    if (idx >= total) return;
    int l = idx % L;
    int t = (idx / L) % T;
    int c = idx / (L * T);
    out[((size_t)t * 3 + c) * L + l] = in[idx];
}

// in [R, C] f32 -> out [C, R] f32  (Whh transpose for coalesced scan reads)
__global__ __launch_bounds__(256) void transpose_mat(
    const float* __restrict__ in, float* __restrict__ out, int R, int C)
{
    int idx = blockIdx.x * blockDim.x + threadIdx.x;
    if (idx >= R * C) return;
    int c = idx % C;
    int r = idx / C;
    out[(size_t)c * R + r] = in[idx];
}

// one-shot f32 -> f16 conversion (GEMM weights)
__global__ __launch_bounds__(256) void cvt_f32_f16(
    const float* __restrict__ in, _Float16* __restrict__ out, int n)
{
    int idx = blockIdx.x * blockDim.x + threadIdx.x;
    if (idx < n) out[idx] = (_Float16)in[idx];
}

// ---------------------------------------------------------------------------
// WMMA GEMM:  C[M, N] = act(A[M,K] @ W[N,K]^T + b0 + b1)
// A, W pre-converted row-major f16; f32 accumulation via
// v_wmma_f32_16x16x32_f16. One 16x32 C-tile per wave (A fragment reused by two
// back-to-back WMMAs), 8 waves / block. N must be a multiple of 32.
// CDNA5 operand layout (ISA 7.12.2):
//   A/B frag: lanes 0..15 -> row(M)/col(N) = lane, k-groups {0..7, 16..23};
//             lanes 16..31 -> row = lane-16, k-groups {8..15, 24..31}.
//   C frag:   VGPR r -> M = r + 8*(lane>=16), N = lane & 15.
// Optional outputs: f32 C and/or f16 C16 (operand for the next GEMM).
// ---------------------------------------------------------------------------
#define CAT16 0,1,2,3,4,5,6,7,8,9,10,11,12,13,14,15

__global__ __launch_bounds__(256) void gemm_wmma_f16(
    const _Float16* __restrict__ A, int lda,
    const _Float16* __restrict__ W, int ldw,
    const float* __restrict__ bias0, const float* __restrict__ bias1,
    float* __restrict__ C, _Float16* __restrict__ C16,
    int ldc, int col_off, int M, int N, int K, int act)
{
    const int lane    = threadIdx.x & 31;
    const int wave    = threadIdx.x >> 5;
    const int tilesN  = N >> 5;                       // 16x32 tiles along N
    const int tile    = blockIdx.x * 8 + wave;
    if (tile >= (M >> 4) * tilesN) return;            // wave-uniform (EXEC all-1)
    const int tm   = tile / tilesN;
    const int tnb  = (tile % tilesN) * 32;
    const int half = lane >> 4;
    const int lr   = lane & 15;

    const _Float16* Ap  = A + (size_t)(tm * 16 + lr) * lda;
    const _Float16* Bp0 = W + (size_t)(tnb + lr) * ldw;
    const _Float16* Bp1 = W + (size_t)(tnb + 16 + lr) * ldw;

    v8f acc0 = {}, acc1 = {};
    int k0 = 0;
    for (; k0 + 32 <= K; k0 += 32) {
        const int kb = k0 + half * 8;
        v8h alo  = *(const v8h*)(Ap  + kb);
        v8h ahi  = *(const v8h*)(Ap  + kb + 16);
        v8h b0lo = *(const v8h*)(Bp0 + kb);
        v8h b0hi = *(const v8h*)(Bp0 + kb + 16);
        v8h b1lo = *(const v8h*)(Bp1 + kb);
        v8h b1hi = *(const v8h*)(Bp1 + kb + 16);
        v16h av  = __builtin_shufflevector(alo,  ahi,  CAT16);
        v16h bv0 = __builtin_shufflevector(b0lo, b0hi, CAT16);
        v16h bv1 = __builtin_shufflevector(b1lo, b1hi, CAT16);
        acc0 = __builtin_amdgcn_wmma_f32_16x16x32_f16(false, av, false, bv0,
                                                      (short)0, acc0, false, false);
        acc1 = __builtin_amdgcn_wmma_f32_16x16x32_f16(false, av, false, bv1,
                                                      (short)0, acc1, false, false);
    }
    if (k0 < K) {   // K-tail (K = 360, 312): zero-padded fragment
        const int kb = k0 + half * 8;
        v16h av = {}, bv0 = {}, bv1 = {};
        for (int i = 0; i < 8; i++) {
            int k1 = kb + i, k2 = kb + 16 + i;
            if (k1 < K) { av[i]   = Ap[k1]; bv0[i]   = Bp0[k1]; bv1[i]   = Bp1[k1]; }
            if (k2 < K) { av[i+8] = Ap[k2]; bv0[i+8] = Bp0[k2]; bv1[i+8] = Bp1[k2]; }
        }
        acc0 = __builtin_amdgcn_wmma_f32_16x16x32_f16(false, av, false, bv0,
                                                      (short)0, acc0, false, false);
        acc1 = __builtin_amdgcn_wmma_f32_16x16x32_f16(false, av, false, bv1,
                                                      (short)0, acc1, false, false);
    }

    const int n0 = tnb + lr;
    const int n1 = tnb + 16 + lr;
    float bs0 = 0.f, bs1 = 0.f;
    if (bias0) { bs0 += bias0[n0]; bs1 += bias0[n1]; }
    if (bias1) { bs0 += bias1[n0]; bs1 += bias1[n1]; }
    #pragma unroll
    for (int r = 0; r < 8; r++) {
        int m = tm * 16 + half * 8 + r;
        float v0 = acc0[r] + bs0;
        float v1 = acc1[r] + bs1;
        if (act == 1) { v0 = fmaxf(v0, 0.f); v1 = fmaxf(v1, 0.f); }
        size_t base = (size_t)m * ldc + col_off;
        if (C)   { C[base + n0]   = v0;            C[base + n1]   = v1; }
        if (C16) { C16[base + n0] = (_Float16)v0;  C16[base + n1] = (_Float16)v1; }
    }
}

// ---------------------------------------------------------------------------
// Sequential LSTM scan (torch gate order i,f,g,o). One 1024-thread workgroup
// (32 waves on one WGP). zx: [T,1024] = x@Wih^T + bih + bhh ; whhT: [256,1024]
// transposed for coalesced reads (1 MB -> resident in L2/WGP$ across steps).
// Outputs: hseq f32 [T,256] and optional f16 copy (next GEMM's operand).
// ---------------------------------------------------------------------------
__device__ __forceinline__ float sigmoidf_(float x) { return 1.f / (1.f + expf(-x)); }

__global__ __launch_bounds__(1024) void lstm_scan(
    const float* __restrict__ zx, const float* __restrict__ whhT,
    float* __restrict__ hseq, _Float16* __restrict__ hseq16, int T)
{
    __shared__ float hb[256];
    __shared__ float zb[1024];
    const int j = threadIdx.x;
    float c = 0.f;
    if (j < 256) hb[j] = 0.f;
    __syncthreads();

    for (int t = 0; t < T; t++) {
        float acc = zx[(size_t)t * 1024 + j];
        #pragma unroll 4
        for (int k = 0; k < 256; k++)
            acc += hb[k] * whhT[(size_t)k * 1024 + j];
        zb[j] = acc;
        __syncthreads();
        if (j < 256) {
            float i = sigmoidf_(zb[j]);
            float f = sigmoidf_(zb[256 + j]);
            float g = tanhf(zb[512 + j]);
            float o = sigmoidf_(zb[768 + j]);
            c = f * c + i * g;
            float h = o * tanhf(c);
            hb[j] = h;
            hseq[(size_t)t * 256 + j] = h;
            if (hseq16) hseq16[(size_t)t * 256 + j] = (_Float16)h;
        }
        __syncthreads();
    }
}

// out[n] = tanh(dot(h, w[n,:]) + b[n])
__global__ __launch_bounds__(128) void head_tanh(
    const float* __restrict__ h, const float* __restrict__ w,
    const float* __restrict__ b, float* __restrict__ out, int N, int H)
{
    int n = blockIdx.x * blockDim.x + threadIdx.x;
    if (n >= N) return;
    float acc = b[n];
    #pragma unroll 4
    for (int k = 0; k < H; k++) acc += h[k] * w[(size_t)n * H + k];
    out[n] = tanhf(acc);
}

// ---------------------------------------------------------------------------
// Host orchestration
// ---------------------------------------------------------------------------
static inline int cdiv(long long a, int b) { return (int)((a + b - 1) / b); }

extern "C" void kernel_launch(void* const* d_in, const int* in_sizes, int n_in,
                              void* d_out, int out_size, void* d_ws, size_t ws_size,
                              hipStream_t stream)
{
    (void)in_sizes; (void)n_in; (void)out_size; (void)ws_size;
    const float* AE     = (const float*)d_in[0];
    const float* Mic    = (const float*)d_in[1];
    const float* Forces = (const float*)d_in[2];
    // d_in[3] = tillTimeT (unused; T fixed by shapes)
    auto prm = [&](int i) { return (const float*)d_in[4 + i]; };

    const float *ae_c1_w=prm(0),  *ae_c1_b=prm(1);
    const float *ae_bn1_g=prm(2), *ae_bn1_b=prm(3), *ae_bn1_m=prm(4), *ae_bn1_v=prm(5);
    const float *ae_c2_w=prm(6),  *ae_c2_b=prm(7);
    const float *ae_bn2_g=prm(8), *ae_bn2_b=prm(9), *ae_bn2_m=prm(10), *ae_bn2_v=prm(11);
    const float *ae_c3_w=prm(12), *ae_c3_b=prm(13);
    const float *ae_d_w=prm(14),  *ae_d_b=prm(15);
    const float *mic_c1_w=prm(16), *mic_c1_b=prm(17);
    const float *mic_bn1_g=prm(18),*mic_bn1_b=prm(19),*mic_bn1_m=prm(20),*mic_bn1_v=prm(21);
    const float *mic_c2_w=prm(22), *mic_c2_b=prm(23);
    const float *mic_bn2_g=prm(24),*mic_bn2_b=prm(25),*mic_bn2_m=prm(26),*mic_bn2_v=prm(27);
    const float *mic_c3_w=prm(28), *mic_c3_b=prm(29);
    const float *mic_d_w=prm(30),  *mic_d_b=prm(31);
    const float *f_c1_w=prm(32), *f_c1_b=prm(33);
    const float *f_bn1_g=prm(34),*f_bn1_b=prm(35),*f_bn1_m=prm(36),*f_bn1_v=prm(37);
    const float *f_c2_w=prm(38), *f_c2_b=prm(39);
    const float *f_bn2_g=prm(40),*f_bn2_b=prm(41),*f_bn2_m=prm(42),*f_bn2_v=prm(43);
    const float *f_c3_w=prm(44), *f_c3_b=prm(45);
    const float *f_d_w=prm(46),  *f_d_b=prm(47);
    const float *l0_wih=prm(48), *l0_whh=prm(49), *l0_bih=prm(50), *l0_bhh=prm(51);
    const float *l1_wih=prm(52), *l1_whh=prm(53), *l1_bih=prm(54), *l1_bhh=prm(55);
    const float *d1_w=prm(56),   *d1_b=prm(57);

    const int T = 1024;

    // workspace carve-up (byte-based, 256B aligned slots)
    char* ws = (char*)d_ws;
    size_t off = 0;
    auto takeF = [&](size_t n) { float* p = (float*)(ws + off);
                                 off += ((n * 4) + 255) & ~(size_t)255; return p; };
    auto takeH = [&](size_t n) { _Float16* p = (_Float16*)(ws + off);
                                 off += ((n * 2) + 255) & ~(size_t)255; return p; };
    float*    bufA   = takeF(5753856);   // max conv scratch (AE conv1: 1024*3*1873)
    float*    bufB   = takeF(1048576);
    float*    zx     = takeF(1048576);   // [1024,1024] input projection (reused)
    float*    h0     = takeF(262144);    // [1024,256] f32
    float*    h1     = takeF(262144);
    float*    whh0T  = takeF(262144);    // [256,1024]
    float*    whh1T  = takeF(262144);
    _Float16* flat16 = takeH(368640);    // pooled features, f16 (<=1024*360)
    _Float16* feat16 = takeH(196608);    // [1024,192] f16
    _Float16* h0_16  = takeH(262144);    // [1024,256] f16
    _Float16* aedw16 = takeH(64 * 360);
    _Float16* micdw16= takeH(64 * 360);
    _Float16* fdw16  = takeH(64 * 312);
    _Float16* wih016 = takeH(1024 * 192);
    _Float16* wih116 = takeH(1024 * 256);

    const int BT = 256;
    #define GRID(n) dim3(cdiv((n), BT)), dim3(BT), 0, stream
    #define GEMM_GRID(M, N) dim3(cdiv(((M) / 16) * ((N) / 32), 8)), dim3(256), 0, stream

    // ---- one-shot weight preps (transposes + f16 conversions) ----
    transpose_mat<<<GRID(1024 * 256)>>>(l0_whh, whh0T, 1024, 256);
    transpose_mat<<<GRID(1024 * 256)>>>(l1_whh, whh1T, 1024, 256);
    cvt_f32_f16<<<GRID(64 * 360)>>>(ae_d_w,  aedw16,  64 * 360);
    cvt_f32_f16<<<GRID(64 * 360)>>>(mic_d_w, micdw16, 64 * 360);
    cvt_f32_f16<<<GRID(64 * 312)>>>(f_d_w,   fdw16,   64 * 312);
    cvt_f32_f16<<<GRID(1024 * 192)>>>(l0_wih, wih016, 1024 * 192);
    cvt_f32_f16<<<GRID(1024 * 256)>>>(l1_wih, wih116, 1024 * 256);

    // ---- AE stack: L 37501 ->1873 ->186 ->92 ->pool 45 ; flat 360 ----
    conv1d_fused<<<GRID(T * 3 * 1873)>>>(AE, ae_c1_w, ae_c1_b,
        ae_bn1_g, ae_bn1_b, ae_bn1_m, ae_bn1_v, bufA, T, 1, 37501, 3, 61, 20, 1873, 0);
    conv1d_fused<<<GRID(T * 5 * 186)>>>(bufA, ae_c2_w, ae_c2_b,
        ae_bn2_g, ae_bn2_b, ae_bn2_m, ae_bn2_v, bufB, T, 3, 1873, 5, 23, 10, 186, 1);
    conv1d_fused<<<GRID(T * 8 * 92)>>>(bufB, ae_c3_w, ae_c3_b,
        nullptr, nullptr, nullptr, nullptr, bufA, T, 5, 186, 8, 4, 2, 92, 2);
    avgpool_flatten_f16<<<GRID(T * 8 * 45)>>>(bufA, flat16, T, 8, 92, 45);
    gemm_wmma_f16<<<GEMM_GRID(1024, 64)>>>(flat16, 360, aedw16, 360, ae_d_b, nullptr,
        nullptr, feat16, 192, 0, 1024, 64, 360, 1);

    // ---- Mic stack: L 9376 ->934 ->185 ->92 ->pool 45 ; flat 360 ----
    conv1d_fused<<<GRID(T * 3 * 934)>>>(Mic, mic_c1_w, mic_c1_b,
        mic_bn1_g, mic_bn1_b, mic_bn1_m, mic_bn1_v, bufA, T, 1, 9376, 3, 46, 10, 934, 0);
    conv1d_fused<<<GRID(T * 5 * 185)>>>(bufA, mic_c2_w, mic_c2_b,
        mic_bn2_g, mic_bn2_b, mic_bn2_m, mic_bn2_v, bufB, T, 3, 934, 5, 14, 5, 185, 1);
    conv1d_fused<<<GRID(T * 8 * 92)>>>(bufB, mic_c3_w, mic_c3_b,
        nullptr, nullptr, nullptr, nullptr, bufA, T, 5, 185, 8, 3, 2, 92, 2);
    avgpool_flatten_f16<<<GRID(T * 8 * 45)>>>(bufA, flat16, T, 8, 92, 45);
    gemm_wmma_f16<<<GEMM_GRID(1024, 64)>>>(flat16, 360, micdw16, 360, mic_d_b, nullptr,
        nullptr, feat16, 192, 64, 1024, 64, 360, 1);

    // ---- Forces stack: [3,T,125]->[T,3,125]; L 125 ->60 ->58 ->28 ->pool 13 ; flat 312
    transpose_forces<<<GRID(3 * T * 125)>>>(Forces, bufA, T, 125);
    conv1d_fused<<<GRID(T * 6 * 60)>>>(bufA, f_c1_w, f_c1_b,
        f_bn1_g, f_bn1_b, f_bn1_m, f_bn1_v, bufB, T, 3, 125, 6, 7, 2, 60, 0);
    conv1d_fused<<<GRID(T * 15 * 58)>>>(bufB, f_c2_w, f_c2_b,
        f_bn2_g, f_bn2_b, f_bn2_m, f_bn2_v, bufA, T, 6, 60, 15, 3, 1, 58, 1);
    conv1d_fused<<<GRID(T * 24 * 28)>>>(bufA, f_c3_w, f_c3_b,
        nullptr, nullptr, nullptr, nullptr, bufB, T, 15, 58, 24, 4, 2, 28, 2);
    avgpool_flatten_f16<<<GRID(T * 24 * 13)>>>(bufB, flat16, T, 24, 28, 13);
    gemm_wmma_f16<<<GEMM_GRID(1024, 64)>>>(flat16, 312, fdw16, 312, f_d_b, nullptr,
        nullptr, feat16, 192, 128, 1024, 64, 312, 1);

    // ---- LSTM layer 0: zx = feat @ Wih0^T + bih0 + bhh0 ; scan ----
    gemm_wmma_f16<<<GEMM_GRID(1024, 1024)>>>(feat16, 192, wih016, 192, l0_bih, l0_bhh,
        zx, nullptr, 1024, 0, 1024, 1024, 192, 0);
    lstm_scan<<<dim3(1), dim3(1024), 0, stream>>>(zx, whh0T, h0, h0_16, T);

    // ---- LSTM layer 1 ----
    gemm_wmma_f16<<<GEMM_GRID(1024, 1024)>>>(h0_16, 256, wih116, 256, l1_bih, l1_bhh,
        zx, nullptr, 1024, 0, 1024, 1024, 256, 0);
    lstm_scan<<<dim3(1), dim3(1024), 0, stream>>>(zx, whh1T, h1, nullptr, T);

    // ---- output head: tanh(h_last @ d1_w^T + d1_b) -> [125] ----
    head_tanh<<<dim3(1), dim3(128), 0, stream>>>(
        h1 + (size_t)(T - 1) * 256, d1_w, d1_b, (float*)d_out, 125, 256);

    #undef GRID
    #undef GEMM_GRID
}
